// BaseLayerWithLoRA_63196148793993
// MI455X (gfx1250) — matrix operand order repaired and checked by reading, not compiled, not run on hardware
//
#include <hip/hip_runtime.h>

// ---------------------------------------------------------------------------
// LoRA linear:  out = x @ W^T + b + (x @ A^T) @ B^T
//   M = B*S = 8192, K = I = 4096, N = O = 4096, R = 16  (all f32)
// Compute-bound (275 GFLOP vs ~14us HBM floor) -> bf16 WMMA pipes with a
// 3-term bf16 split (hi*hi + hi*lo + lo*hi) for ~f32 accuracy at bf16 rate/3.
// hi = RNE bf16, lo = truncated residual (error ~2^-17, packing is free).
// ---------------------------------------------------------------------------

#define M_TOT 8192
#define K_TOT 4096
#define N_TOT 4096
#define R_LORA 16

#define BM 128
#define BN 128
#define KC 32
#define LDSROW 40   // bf16 elements per LDS row: 32 data + 8 pad (80B stride)

typedef __attribute__((ext_vector_type(16))) __bf16     v16bf;
typedef __attribute__((ext_vector_type(2)))  __bf16     v2bf;
typedef __attribute__((ext_vector_type(8)))  float      v8f;
typedef __attribute__((ext_vector_type(4)))  float      f32x4;
typedef __attribute__((ext_vector_type(2)))  float      f32x2;
typedef __attribute__((ext_vector_type(4)))  unsigned   u32x4;
typedef __attribute__((ext_vector_type(2)))  unsigned   u32x2;

struct Pack8  { u32x2 a, b, c, d; };   // 32 B -> v16bf
struct Pack16 { u32x4 a, b;       };   // 32 B -> v16bf

// ---- bf16 split helpers ----------------------------------------------------
#if __has_builtin(__builtin_amdgcn_cvt_pk_bf16_f32)
// packed = {bf16(a) in [15:0], bf16(b) in [31:16]}, RNE in hardware
__device__ __forceinline__ unsigned pk_bf16(float a, float b) {
  return __builtin_bit_cast(unsigned, __builtin_amdgcn_cvt_pk_bf16_f32(a, b));
}
__device__ __forceinline__ void cvt_split2(float a, float b,
                                           unsigned* hi, unsigned* lo) {
  unsigned h = pk_bf16(a, b);
  float ha = __builtin_bit_cast(float, h << 16);
  float hb = __builtin_bit_cast(float, h & 0xffff0000u);
  *hi = h;
  *lo = pk_bf16(a - ha, b - hb);
}
#else
// Manual path. hi: RNE (bfe+add3+and per element). lo: residual bits truncated
// directly from the f32 pattern -> packing costs 2 ops per PAIR, no per-elem op.
__device__ __forceinline__ unsigned rne_hibits(float f) {
  unsigned u = __builtin_bit_cast(unsigned, f);
  return (u + 0x7FFFu + ((u >> 16) & 1u)) & 0xffff0000u;
}
__device__ __forceinline__ void cvt_split2(float a, float b,
                                           unsigned* hi, unsigned* lo) {
  unsigned ha = rne_hibits(a), hb = rne_hibits(b);
  *hi = (ha >> 16) | hb;
  float ra = a - __builtin_bit_cast(float, ha);
  float rb = b - __builtin_bit_cast(float, hb);
  unsigned la = __builtin_bit_cast(unsigned, ra);
  unsigned lb = __builtin_bit_cast(unsigned, rb);
  *lo = (la >> 16) | (lb & 0xffff0000u);   // truncated residual bf16s
}
#endif

__device__ __forceinline__ void cvt_split4(f32x4 v, u32x2* hi, u32x2* lo) {
  unsigned h0, l0, h1, l1;
  cvt_split2(v.x, v.y, &h0, &l0);
  cvt_split2(v.z, v.w, &h1, &l1);
  u32x2 h, l;
  h.x = h0; h.y = h1;
  l.x = l0; l.y = l1;
  *hi = h;  *lo = l;
}

__device__ __forceinline__ v16bf frag_from(u32x2 a, u32x2 b, u32x2 c, u32x2 d) {
  Pack8 p{a, b, c, d};
  return __builtin_bit_cast(v16bf, p);
}
__device__ __forceinline__ v16bf frag_lds(const unsigned short* p0,
                                          const unsigned short* p1) {
  Pack16 p;
  p.a = *(const u32x4*)p0;   // ds_load_b128
  p.b = *(const u32x4*)p1;   // ds_load_b128
  return __builtin_bit_cast(v16bf, p);
}

__device__ __forceinline__ v8f wmma_bf16(v16bf a, v16bf b, v8f c) {
  return __builtin_amdgcn_wmma_f32_16x16x32_bf16(
      /*neg_a=*/false, a, /*neg_b=*/false, b,
      /*c_mod=*/(short)0, c, /*reuse_a=*/false, /*reuse_b=*/false);
}

__device__ __forceinline__ float dot4(f32x4 a, f32x4 b) {
  return a.x * b.x + a.y * b.y + a.z * b.z + a.w * b.w;
}

// ---------------------------------------------------------------------------
// Kernel 1: AR[M,16] = x @ lora_A^T  (one 16x16 WMMA tile per wave)
// lora_A tile (shared by all 8 waves) staged through LDS, converted once.
// x fragments loaded per-lane directly (K-contiguous rows).
// ---------------------------------------------------------------------------
__global__ __launch_bounds__(256)
void lora_ar_kernel(const float* __restrict__ x,
                    const float* __restrict__ loraA,
                    float* __restrict__ AR) {
  __shared__ __align__(16) unsigned short sBh[16 * LDSROW];
  __shared__ __align__(16) unsigned short sBl[16 * LDSROW];

  const int t    = threadIdx.x;
  const int wave = t >> 5;
  const int lane = t & 31;
  const int hlf  = lane >> 4;      // 0: lanes 0-15, 1: lanes 16-31
  const int l16  = lane & 15;
  const int m0   = blockIdx.x * 128 + wave * 16;

  v8f C = {0.f, 0.f, 0.f, 0.f, 0.f, 0.f, 0.f, 0.f};

  const float* xrow = x + (size_t)(m0 + l16) * K_TOT;   // A: row M=l16
  const int o0 = hlf * 8;  // A half K offset

  // staging map for the 16x32 lora_A tile: 512 f32, 2 per thread
  const int brow = t >> 4;         // 0..15
  const int bq   = t & 15;         // pair index, k offset 2*bq

  for (int k0 = 0; k0 < K_TOT; k0 += KC) {
    // ---- stage lora_A tile -> bf16 hi/lo planes (converted once/block) ----
    {
      f32x2 v = *(const f32x2*)(loraA + (size_t)brow * K_TOT + k0 + bq * 2);
      unsigned h, l;
      cvt_split2(v.x, v.y, &h, &l);
      *(unsigned*)(sBh + brow * LDSROW + bq * 2) = h;
      *(unsigned*)(sBl + brow * LDSROW + bq * 2) = l;
    }

    // ---- A fragment from global: K {0-7,16-23} / {8-15,24-31} per half ----
    f32x4 a0 = *(const f32x4*)(xrow + k0 + o0);
    f32x4 a1 = *(const f32x4*)(xrow + k0 + o0 + 4);
    f32x4 a2 = *(const f32x4*)(xrow + k0 + o0 + 16);
    f32x4 a3 = *(const f32x4*)(xrow + k0 + o0 + 20);
    u32x2 ah0, ah1, ah2, ah3, al0, al1, al2, al3;
    cvt_split4(a0, &ah0, &al0);  cvt_split4(a1, &ah1, &al1);
    cvt_split4(a2, &ah2, &al2);  cvt_split4(a3, &ah3, &al3);
    v16bf ah = frag_from(ah0, ah1, ah2, ah3);
    v16bf al = frag_from(al0, al1, al2, al3);

    __syncthreads();
    // ---- B fragment from LDS (32x16): K {0-7,16-23} / {8-15,24-31} per half ----
    {
      int rb = l16 * LDSROW + hlf * 8;
      v16bf bh = frag_lds(sBh + rb, sBh + rb + 16);
      v16bf bl = frag_lds(sBl + rb, sBl + rb + 16);
      C = wmma_bf16(ah, bh, C);
      C = wmma_bf16(ah, bl, C);
      C = wmma_bf16(al, bh, C);
    }
    __syncthreads();
  }

#pragma unroll
  for (int v = 0; v < 8; ++v) {
    int m = m0 + hlf * 8 + v;      // C layout: VGPR v -> M = v + 8*half
    AR[(size_t)m * R_LORA + l16] = C[v];
  }
}

// ---------------------------------------------------------------------------
// Kernel 2: out = x @ W^T + b + AR @ lora_B^T
// 128x128 block tile, 8 waves (2Mx4N), 64x32 per wave, KC=32 chunks.
// f32->bf16 hi/lo split happens once during global->LDS staging.
// ---------------------------------------------------------------------------
__global__ __launch_bounds__(256)
void lora_gemm_kernel(const float* __restrict__ x,
                      const float* __restrict__ W,
                      const float* __restrict__ bvec,
                      const float* __restrict__ loraB,
                      const float* __restrict__ AR,
                      float* __restrict__ out) {
  __shared__ __align__(16) unsigned short sAhi[BM * LDSROW];
  __shared__ __align__(16) unsigned short sAlo[BM * LDSROW];
  __shared__ __align__(16) unsigned short sBhi[BN * LDSROW];
  __shared__ __align__(16) unsigned short sBlo[BN * LDSROW];

  const int t    = threadIdx.x;
  const int wave = t >> 5;
  const int lane = t & 31;
  const int wm   = wave >> 2;      // 0..1  (64 rows each)
  const int wn   = wave & 3;       // 0..3  (32 cols each)
  const int hlf  = lane >> 4;
  const int l16  = lane & 15;

  const int blockN = blockIdx.x * BN;   // N fast -> x rows reused via L2
  const int blockM = blockIdx.y * BM;

  v8f C[4][2];
#pragma unroll
  for (int mt = 0; mt < 4; ++mt)
#pragma unroll
    for (int nt = 0; nt < 2; ++nt)
      C[mt][nt] = (v8f){0.f, 0.f, 0.f, 0.f, 0.f, 0.f, 0.f, 0.f};

  // Staging map: idx = i*256+t, row = idx>>3, q = idx&7 (8 float4 per row)
  const int srow = t >> 3;        // base row for i=0
  const int sq   = t & 7;

  for (int k0 = 0; k0 < K_TOT; k0 += KC) {
    if (k0 + KC < K_TOT) {  // prefetch next chunk (global_prefetch_b8)
      __builtin_prefetch(x + (size_t)(blockM + srow) * K_TOT + k0 + KC + sq * 4, 0, 0);
      __builtin_prefetch(W + (size_t)(blockN + srow) * K_TOT + k0 + KC + sq * 4, 0, 0);
    }

    // ---- stage x tile (128 x 32 f32 -> bf16 hi/lo planes) ----
#pragma unroll
    for (int i = 0; i < 4; ++i) {
      int idx = i * 256 + t;
      int row = idx >> 3, q = idx & 7;
      f32x4 v = *(const f32x4*)(x + (size_t)(blockM + row) * K_TOT + k0 + q * 4);
      u32x2 h, l;
      cvt_split4(v, &h, &l);
      *(u32x2*)(sAhi + row * LDSROW + q * 4) = h;
      *(u32x2*)(sAlo + row * LDSROW + q * 4) = l;
    }
    // ---- stage W tile ----
#pragma unroll
    for (int i = 0; i < 4; ++i) {
      int idx = i * 256 + t;
      int row = idx >> 3, q = idx & 7;
      f32x4 v = *(const f32x4*)(W + (size_t)(blockN + row) * K_TOT + k0 + q * 4);
      u32x2 h, l;
      cvt_split4(v, &h, &l);
      *(u32x2*)(sBhi + row * LDSROW + q * 4) = h;
      *(u32x2*)(sBlo + row * LDSROW + q * 4) = l;
    }
    __syncthreads();

    // ---- load fragments from LDS ----
    v16bf ah[4], al[4], bh[2], bl[2];
#pragma unroll
    for (int mt = 0; mt < 4; ++mt) {
      int rb = (wm * 64 + mt * 16 + l16) * LDSROW;
      int o0 = hlf * 8;            // A: K {0-7,16-23} vs {8-15,24-31}
      int o1 = 16 + hlf * 8;
      ah[mt] = frag_lds(sAhi + rb + o0, sAhi + rb + o1);
      al[mt] = frag_lds(sAlo + rb + o0, sAlo + rb + o1);
    }
#pragma unroll
    for (int nt = 0; nt < 2; ++nt) {
      int rb = (wn * 32 + nt * 16 + l16) * LDSROW + hlf * 8; // B: K {0-7,16-23} vs {8-15,24-31}
      bh[nt] = frag_lds(sBhi + rb, sBhi + rb + 16);
      bl[nt] = frag_lds(sBlo + rb, sBlo + rb + 16);
    }

    // ---- 8 tiles x 3 WMMAs (hi*hi + hi*lo + lo*hi) ----
#pragma unroll
    for (int mt = 0; mt < 4; ++mt)
#pragma unroll
      for (int nt = 0; nt < 2; ++nt) {
        C[mt][nt] = wmma_bf16(ah[mt], bh[nt], C[mt][nt]);
        C[mt][nt] = wmma_bf16(ah[mt], bl[nt], C[mt][nt]);
        C[mt][nt] = wmma_bf16(al[mt], bh[nt], C[mt][nt]);
      }
    __syncthreads();
  }

  // ---- epilogue: + bias + AR @ lora_B^T (exact f32, rank 16) ----
  int   ncol[2];
  float bias[2];
  f32x4 lb[2][4];
#pragma unroll
  for (int nt = 0; nt < 2; ++nt) {
    int n = blockN + wn * 32 + nt * 16 + l16;
    ncol[nt] = n;
    bias[nt] = bvec[n];
    const f32x4* lp = (const f32x4*)(loraB + (size_t)n * R_LORA);
    lb[nt][0] = lp[0]; lb[nt][1] = lp[1]; lb[nt][2] = lp[2]; lb[nt][3] = lp[3];
  }
#pragma unroll
  for (int mt = 0; mt < 4; ++mt) {
#pragma unroll
    for (int v = 0; v < 8; ++v) {
      int m = blockM + wm * 64 + mt * 16 + hlf * 8 + v;
      const f32x4* ap = (const f32x4*)(AR + (size_t)m * R_LORA);
      f32x4 a0 = ap[0], a1 = ap[1], a2 = ap[2], a3 = ap[3];
#pragma unroll
      for (int nt = 0; nt < 2; ++nt) {
        float d = dot4(a0, lb[nt][0]) + dot4(a1, lb[nt][1]) +
                  dot4(a2, lb[nt][2]) + dot4(a3, lb[nt][3]);
        out[(size_t)m * N_TOT + ncol[nt]] = C[mt][nt][v] + d + bias[nt];
      }
    }
  }
}

// ---------------------------------------------------------------------------
extern "C" void kernel_launch(void* const* d_in, const int* in_sizes, int n_in,
                              void* d_out, int out_size, void* d_ws,
                              size_t ws_size, hipStream_t stream) {
  (void)in_sizes; (void)n_in; (void)out_size; (void)ws_size;
  const float* x      = (const float*)d_in[0];
  const float* W      = (const float*)d_in[1];
  const float* b      = (const float*)d_in[2];
  const float* lora_A = (const float*)d_in[3];
  const float* lora_B = (const float*)d_in[4];
  float* out = (float*)d_out;
  float* AR  = (float*)d_ws;   // 8192 x 16 f32 = 512 KB scratch

  lora_ar_kernel<<<dim3(M_TOT / 128), dim3(256), 0, stream>>>(x, lora_A, AR);
  lora_gemm_kernel<<<dim3(N_TOT / BN, M_TOT / BM), dim3(256), 0, stream>>>(
      x, W, b, lora_B, AR, out);
}
